// EpiAttn_54786602828447
// MI455X (gfx1250) — matrix operand order repaired and checked by reading, not compiled, not run on hardware
//
#include <hip/hip_runtime.h>
#include <hip/hip_bf16.h>
#include <math.h>

typedef __attribute__((ext_vector_type(16))) __bf16 v16bf;
typedef __attribute__((ext_vector_type(8)))  float  v8f;

#define BB 256
#define LL 512
#define FF 28
#define HH 512
#define NWG 32
#define LTPB 512

// workspace byte offsets
#define OFF_BAR   0u
#define OFF_WA    4096u        // 256 f32
#define OFF_CTX   8192u        // 256*32 bf16 = 16KB
#define OFF_HB    32768u       // 2 * 256*512 bf16 = 512KB
#define OFF_P     589824u      // 256*512 f32 = 512KB   (normalized attn, [l][b])
#define OFF_UA    1114112u     // 512KB  ([b][l])
#define OFF_BIAS  1638400u     // 512KB  ([b][l])
#define OFF_XF    2162688u     // 14680064 ([b][l][f])
#define OFF_PART  16842752u    // 512*256*32 f32 = 16MB ([l][b][wg])

__device__ inline unsigned short f2bf(float f) {
  unsigned u = __float_as_uint(f);
  unsigned r = u + 0x7FFFu + ((u >> 16) & 1u);
  return (unsigned short)(r >> 16);
}
__device__ inline float bf2f(unsigned short h) {
  return __uint_as_float(((unsigned)h) << 16);
}
__device__ inline unsigned packbf2(float lo, float hi) {
  return (unsigned)f2bf(lo) | ((unsigned)f2bf(hi) << 16);
}
__device__ inline float sigm(float x) { return 1.f / (1.f + __expf(-x)); }
__device__ inline v8f splat8(float x) { v8f r = {x,x,x,x,x,x,x,x}; return r; }

__device__ inline v8f wmma_bf16(v16bf a, v16bf b, v8f c) {
  return __builtin_amdgcn_wmma_f32_16x16x32_bf16(false, a, false, b, (short)0, c, false, false);
}
// same-A back-to-back WMMA: hint HW to reuse the cached A operand
__device__ inline v8f wmma_bf16_ra(v16bf a, v16bf b, v8f c) {
  return __builtin_amdgcn_wmma_f32_16x16x32_bf16(false, a, false, b, (short)0, c, true, false);
}
__device__ inline v16bf loadA16(const uint4* base, int q0, int q1) {
  union { uint4 q[2]; v16bf v; } u;
  u.q[0] = base[q0];
  u.q[1] = base[q1];
  return u.v;
}
__device__ inline v16bf ldsB(const unsigned* wl, int idx) {
  union { uint4 q[2]; v16bf v; } u;
  u.q[0] = *(const uint4*)&wl[idx];
  u.q[1] = *(const uint4*)&wl[idx + 4];
  return u.v;
}

__device__ inline void grid_barrier(unsigned* cnt, unsigned* gen) {
  __syncthreads();
  if (threadIdx.x == 0) {
    __threadfence();
    unsigned g = __hip_atomic_load(gen, __ATOMIC_RELAXED, __HIP_MEMORY_SCOPE_AGENT);
    unsigned a = __hip_atomic_fetch_add(cnt, 1u, __ATOMIC_ACQ_REL, __HIP_MEMORY_SCOPE_AGENT);
    if (a == (unsigned)(NWG - 1)) {
      __hip_atomic_store(cnt, 0u, __ATOMIC_RELAXED, __HIP_MEMORY_SCOPE_AGENT);
      __hip_atomic_fetch_add(gen, 1u, __ATOMIC_RELEASE, __HIP_MEMORY_SCOPE_AGENT);
    } else {
      while (__hip_atomic_load(gen, __ATOMIC_ACQUIRE, __HIP_MEMORY_SCOPE_AGENT) == g)
        __builtin_amdgcn_s_sleep(1);
    }
    __threadfence();
  }
  __syncthreads();
}

// -------- Kernel 1: conv1 + ELU + residual -> xf ; u_a ; bias_mat ; init state
__global__ void prep_kernel(const float* __restrict__ x, const float* __restrict__ mask,
                            const float* __restrict__ c1w, const float* __restrict__ c1b,
                            const float* __restrict__ cw, const float* __restrict__ cb,
                            char* __restrict__ ws) {
  int tid = blockIdx.x * blockDim.x + threadIdx.x;  // 0..131071
  float* xf = (float*)(ws + OFF_XF);
  float* ua = (float*)(ws + OFF_UA);
  float* bm = (float*)(ws + OFF_BIAS);
  unsigned* hb0 = (unsigned*)(ws + OFF_HB);
  unsigned short* ctx = (unsigned short*)(ws + OFF_CTX);
  unsigned* bar = (unsigned*)(ws + OFF_BAR);

  if (tid == 0) { bar[0] = 0u; bar[1] = 0u; }
  if (tid < (BB * HH / 2)) hb0[tid] = 0u;  // zero h ping buffer (bf16 pairs)
  if (tid >= 65536 && tid < 65536 + BB * 4) {  // zero K-pad cols 28..31 of context
    int idx = tid - 65536;
    ctx[(idx >> 2) * 32 + FF + (idx & 3)] = 0;
  }

  int b = tid / LL, l = tid % LL;
  float xl[3 * FF];
  for (int k = 0; k < 3; ++k) {
    int lp = l + k - 1;
    bool ok = (lp >= 0 && lp < LL);
    for (int i = 0; i < FF; ++i)
      xl[k * FF + i] = ok ? x[(b * LL + lp) * FF + i] : 0.f;
  }
  float m = mask[b * LL + l];
  float uacc = cb[0];
  for (int f = 0; f < FF; ++f) {
    float acc = c1b[f];
    for (int k = 0; k < 3; ++k)
      for (int i = 0; i < FF; ++i)
        acc += xl[k * FF + i] * c1w[f * FF * 3 + i * 3 + k];
    float v = acc * m;
    float e = v > 0.f ? v : (__expf(v) - 1.f);  // ELU
    float xfv = e + xl[FF + f];                 // + residual x[b,l,f]
    xf[(b * LL + l) * FF + f] = xfv;
    uacc += xfv * cw[f];
  }
  ua[b * LL + l] = uacc;
  bm[b * LL + l] = 1e9f * (m - 1.f);
}

// -------- Kernel 2: persistent LSTM+attention, WMMA gates, grid barriers per phase
__global__ void __launch_bounds__(LTPB, 1)
lstm_kernel(const float* __restrict__ Wih, const float* __restrict__ Whh,
            const float* __restrict__ bih, const float* __restrict__ bhh,
            const float* __restrict__ fc1w, const float* __restrict__ fc1b,
            const float* __restrict__ c2w, char* __restrict__ ws) {
  __shared__ unsigned wlds[16384];  // 64KB: W_hh bf16 slab in per-lane WMMA-B layout

  const int wg = blockIdx.x, tix = threadIdx.x;
  const int wv = tix >> 5, lane = tix & 31;
  const int n = lane & 15, hv = lane >> 4;
  const int jcol = wg * 16 + n;

  float* wa = (float*)(ws + OFF_WA);
  float* p = (float*)(ws + OFF_P);
  float* ua = (float*)(ws + OFF_UA);
  float* bm = (float*)(ws + OFF_BIAS);
  float* xf = (float*)(ws + OFF_XF);
  float* part = (float*)(ws + OFF_PART);
  unsigned short* ctx = (unsigned short*)(ws + OFF_CTX);
  unsigned short* hbuf0 = (unsigned short*)(ws + OFF_HB);
  unsigned* bar = (unsigned*)(ws + OFF_BAR);

  // Stage this WG's W_hh slab (4 gates x 16 K-chunks x 16 cols, K=512) into LDS, bf16.
  for (int idx = tix; idx < 16384; idx += LTPB) {
    int d = idx & 7, t = (idx >> 3) & 31, kc = (idx >> 8) & 15, g = idx >> 12;
    int mrow = g * HH + wg * 16 + (t & 15);
    int kb = kc * 32 + (t >> 4) * 16 + 2 * d;
    wlds[idx] = packbf2(Whh[mrow * HH + kb], Whh[mrow * HH + kb + 1]);
  }
  // W_ih B operands (K padded 28 -> 32), held in VGPRs for all 512 steps.
  v16bf wih[4];
  for (int g = 0; g < 4; ++g) {
    union { unsigned u[8]; v16bf v; } t;
    int mrow = g * HH + jcol;
    for (int d = 0; d < 8; ++d) {
      int kb = hv * 16 + 2 * d;
      float lo = (kb < FF) ? Wih[mrow * FF + kb] : 0.f;
      float hi = (kb + 1 < FF) ? Wih[mrow * FF + kb + 1] : 0.f;
      t.u[d] = packbf2(lo, hi);
    }
    wih[g] = t.v;
  }
  float bsv[4];
  for (int g = 0; g < 4; ++g) bsv[g] = bih[g * HH + jcol] + bhh[g * HH + jcol];
  const float w2hj = c2w[jcol];
  const int brow = wv * 16 + n;  // A-tile row (M = lane%16)
  const unsigned* wl = wlds + lane * 8;
  float cst[8];                  // cell state tile, registers for all steps
  for (int r = 0; r < 8; ++r) cst[r] = 0.f;
  __syncthreads();

  for (int l = 0; l < LL; ++l) {
    const unsigned short* hcur = hbuf0 + (l & 1) * (BB * HH);
    unsigned short* hnxt = hbuf0 + ((l & 1) ^ 1) * (BB * HH);
    const uint4* h4 = (const uint4*)hcur;  // 64 uint4 per row

    // ---- Phase A: w_a[b] = h . fc1_w + fc1_b   (WGs 0..15, one b per wave)
    if (wg < 16) {
      int b = wg * 16 + wv;
      float acc = 0.f;
      for (int it = 0; it < 16; ++it) {
        int k = lane + 32 * it;
        acc += bf2f(hcur[b * HH + k]) * fc1w[k];
      }
      for (int mm = 1; mm < 32; mm <<= 1) acc += __shfl_xor(acc, mm, 32);
      if (lane == 0) wa[b] = acc + fc1b[0];
    }
    grid_barrier(bar, bar + 1);

    // ---- Phase B: batch-axis softmax per l-column (one column per wave)
    {
      int lcol = wg * 16 + wv;
      float vals[8];
      float vmax = -3.4e38f;
      for (int i = 0; i < 8; ++i) {
        int b = lane + 32 * i;
        float v = ua[b * LL + lcol] + wa[b];
        v = v > 0.f ? v : 0.01f * v;  // leaky_relu
        v += bm[b * LL + lcol];
        vals[i] = v;
        vmax = fmaxf(vmax, v);
      }
      for (int mm = 1; mm < 32; mm <<= 1) vmax = fmaxf(vmax, __shfl_xor(vmax, mm, 32));
      float s = 0.f;
      for (int i = 0; i < 8; ++i) { vals[i] = __expf(vals[i] - vmax); s += vals[i]; }
      for (int mm = 1; mm < 32; mm <<= 1) s += __shfl_xor(s, mm, 32);
      float inv = 1.f / s;
      for (int i = 0; i < 8; ++i) p[lcol * BB + lane + 32 * i] = vals[i] * inv;
    }
    grid_barrier(bar, bar + 1);

    // ---- Phase C: context[b,f] = sum_l p[l,b] * xf[b,l,f]  -> bf16, K-padded
    //      4 independent accumulators to break the serial FP dependence chain.
    {
      int gt = wg * LTPB + tix;
      if (gt < BB * FF) {
        int b = gt / FF, f = gt - b * FF;
        const float* xrow = xf + (size_t)b * LL * FF + f;
        float acc0 = 0.f, acc1 = 0.f, acc2 = 0.f, acc3 = 0.f;
        for (int l2 = 0; l2 < LL; l2 += 4) {
          acc0 += p[(l2 + 0) * BB + b] * xrow[(l2 + 0) * FF];
          acc1 += p[(l2 + 1) * BB + b] * xrow[(l2 + 1) * FF];
          acc2 += p[(l2 + 2) * BB + b] * xrow[(l2 + 2) * FF];
          acc3 += p[(l2 + 3) * BB + b] * xrow[(l2 + 3) * FF];
        }
        ctx[b * 32 + f] = f2bf((acc0 + acc1) + (acc2 + acc3));
      }
    }
    grid_barrier(bar, bar + 1);

    // ---- Phase D: gates GEMM (WMMA) + LSTM cell
    {
      v8f a0 = splat8(bsv[0]), a1 = splat8(bsv[1]), a2 = splat8(bsv[2]), a3 = splat8(bsv[3]);
      const uint4* c4 = (const uint4*)ctx;  // 4 uint4 per row
      v16bf Ac = loadA16(c4, brow * 4 + 2 * hv, brow * 4 + 2 * hv + 1);
      a0 = wmma_bf16(Ac, wih[0], a0);
      a1 = wmma_bf16_ra(Ac, wih[1], a1);
      a2 = wmma_bf16_ra(Ac, wih[2], a2);
      a3 = wmma_bf16_ra(Ac, wih[3], a3);
      for (int kc = 0; kc < 16; ++kc) {
        v16bf Ah = loadA16(h4, brow * 64 + kc * 4 + 2 * hv, brow * 64 + kc * 4 + 2 * hv + 1);
        if (kc < 15) __builtin_prefetch(&h4[brow * 64 + (kc + 1) * 4 + hv], 0, 1);
        // hoist all four gates' B tiles: lets the scheduler clause 8x ds_load_b128
        // and use staggered s_wait_dscnt instead of full drains per WMMA.
        v16bf B0 = ldsB(wl, (0 * 16 + kc) * 256);
        v16bf B1 = ldsB(wl, (1 * 16 + kc) * 256);
        v16bf B2 = ldsB(wl, (2 * 16 + kc) * 256);
        v16bf B3 = ldsB(wl, (3 * 16 + kc) * 256);
        a0 = wmma_bf16(Ah, B0, a0);
        a1 = wmma_bf16_ra(Ah, B1, a1);
        a2 = wmma_bf16_ra(Ah, B2, a2);
        a3 = wmma_bf16_ra(Ah, B3, a3);
      }
      for (int r = 0; r < 8; ++r) {
        float gi = sigm(a0[r]), gf = sigm(a1[r]), gg = tanhf(a2[r]), go = sigm(a3[r]);
        cst[r] = gf * cst[r] + gi * gg;
        float hn = go * tanhf(cst[r]);
        int b2 = wv * 16 + r + hv * 8;
        hnxt[b2 * HH + jcol] = f2bf(hn);
        float val = hn * w2hj;  // partial of conv2 dot over this j-slab
        for (int mm = 1; mm < 16; mm <<= 1) val += __shfl_xor(val, mm, 32);
        if (n == 0) part[((size_t)l * BB + b2) * NWG + wg] = val;
      }
    }
    grid_barrier(bar, bar + 1);
  }
}

// -------- Kernel 3: reduce slab partials + x-part of conv2, apply mask
__global__ void final_kernel(const float* __restrict__ x, const float* __restrict__ mask,
                             const float* __restrict__ c2w, const float* __restrict__ c2b,
                             const char* __restrict__ ws, float* __restrict__ out) {
  int tid = blockIdx.x * blockDim.x + threadIdx.x;
  int b = tid / LL, l = tid % LL;
  const float* part = (const float*)(ws + OFF_PART);
  const float* pp = part + ((size_t)l * BB + b) * NWG;
  float s = 0.f;
  for (int wg = 0; wg < NWG; ++wg) s += pp[wg];
  float xs = 0.f;
  for (int f = 0; f < FF; ++f) xs += x[(b * LL + l) * FF + f] * c2w[HH + f];
  float m = mask[b * LL + l];
  out[b * LL + l] = m * (m * s + xs + c2b[0]);
}

extern "C" void kernel_launch(void* const* d_in, const int* in_sizes, int n_in,
                              void* d_out, int out_size, void* d_ws, size_t ws_size,
                              hipStream_t stream) {
  (void)in_sizes; (void)n_in; (void)out_size; (void)ws_size;
  const float* x    = (const float*)d_in[0];
  const float* mask = (const float*)d_in[1];
  const float* c1w  = (const float*)d_in[2];
  const float* c1b  = (const float*)d_in[3];
  const float* cw   = (const float*)d_in[4];
  const float* cb   = (const float*)d_in[5];
  const float* c2w  = (const float*)d_in[6];
  const float* c2b  = (const float*)d_in[7];
  const float* Wih  = (const float*)d_in[8];
  const float* Whh  = (const float*)d_in[9];
  const float* bih  = (const float*)d_in[10];
  const float* bhh  = (const float*)d_in[11];
  const float* f1w  = (const float*)d_in[12];
  const float* f1b  = (const float*)d_in[13];
  char* ws = (char*)d_ws;
  float* out = (float*)d_out;

  prep_kernel<<<dim3(BB * LL / 256), dim3(256), 0, stream>>>(x, mask, c1w, c1b, cw, cb, ws);
  lstm_kernel<<<dim3(NWG), dim3(LTPB), 0, stream>>>(Wih, Whh, bih, bhh, f1w, f1b, c2w, ws);
  final_kernel<<<dim3(BB * LL / 256), dim3(256), 0, stream>>>(x, mask, c2w, c2b, ws, out);
}
